// TabMixer_7584912244828
// MI455X (gfx1250) — compile-verified, matrix operand order checked by reading, not run on hardware
//
#include <hip/hip_runtime.h>
#include <hip/hip_bf16.h>
#include <stdint.h>

// ---------------------------------------------------------------------------
// MI455X (gfx1250) implementation.
// Compute-bound (~0.75 TFLOP vs ~10us of HBM traffic) -> all GEMMs run on
// v_wmma_f32_16x16x32_bf16 (wave32, fp32 accum). Weights are transposed +
// converted to bf16 once ([Np][Kp], zero-padded); activations flow as bf16
// wherever they only feed a GEMM. The GEMM inner loop stages tiles with
// 4x b128 global loads + 4x ds_store_b128 per thread -- no converts, no
// guards, no transpose -- software-pipelined against the WMMA stream.
// ---------------------------------------------------------------------------

typedef __attribute__((ext_vector_type(16))) __bf16 v16bf;
typedef __attribute__((ext_vector_type(8)))  __bf16 v8bf;
typedef __attribute__((ext_vector_type(8)))  float  v8f;

#define TILE_M 128
#define TILE_N 128
#define TILE_K 32
#define LDSS   40   // padded LDS row stride (bf16): 80B rows -> conflict-free frag reads

enum { EPI_NONE=0, EPI_IND=1, EPI_GELU_IND=2, EPI_SIG_IND=3, EPI_SIGMOID=4, EPI_MASK_RESID=5 };

#define GELU1 0.8413447460685429f   // gelu(1) = Phi(1)
#define SIG1  0.7310585786300049f   // sigmoid(1)
#define SIG0  0.5f                  // sigmoid(0)

static __host__ __device__ inline int ceil32(int x)  { return (x + 31)  & ~31;  }
static __host__ __device__ inline int ceil128(int x) { return (x + 127) & ~127; }

// ---------------------------------------------------------------------------
// WMMA GEMM: out = epi(A(M x Kp, bf16, zero-padded) @ Wt^T + bias)
// Wt is bf16 [Np][Kp] (row n = weight column n). M always multiple of 128.
// Workgroup = 256 thr (8 waves), tile 128x128x32; wave = 2x4 16x16 subtiles.
// Frag layout (16x32 bf16): lane l: idx=l&15, half=l>>4; two b128 LDS reads
// at k = 8*half and 16+8*half. C/D: VGPR r, lane l -> (M=r+8*(l>>4), N=l&15).
// ---------------------------------------------------------------------------
template<int EPI>
__global__ __launch_bounds__(256)
void wmma_gemm_kernel(const __bf16* __restrict__ A, const __bf16* __restrict__ Wt,
                      const float* __restrict__ bias,
                      float* outf, __bf16* outb,
                      int M, int Kp, int N,
                      const __bf16* __restrict__ mbuf, const float* hbuf)
{
    __shared__ __attribute__((aligned(16))) __bf16 As[TILE_M * LDSS];
    __shared__ __attribute__((aligned(16))) __bf16 Bs[TILE_N * LDSS];

    const int t    = threadIdx.x;
    const int wave = t >> 5;
    const int lane = t & 31;
    const int wm   = wave & 3;        // 4 waves along M
    const int wn   = wave >> 2;       // 2 waves along N
    const int half = lane >> 4;
    const int l16  = lane & 15;
    const int row0 = blockIdx.x * TILE_M;
    const int col0 = blockIdx.y * TILE_N;

    // staging: thread covers row sr = t>>1, k-halfchunk skh = (t&1)*16 (16 bf16 = 32B)
    const int sr  = t >> 1;
    const int skh = (t & 1) * 16;
    const __bf16* gA = A  + (size_t)(row0 + sr) * Kp + skh;
    const __bf16* gB = Wt + (size_t)(col0 + sr) * Kp + skh;   // col0+sr < Np always
    __bf16* sA = &As[sr * LDSS + skh];
    __bf16* sB = &Bs[sr * LDSS + skh];

    v8bf pa0, pa1, pb0, pb1;
    auto loadTiles = [&](int k0) {
        pa0 = *reinterpret_cast<const v8bf*>(gA + k0);
        pa1 = *reinterpret_cast<const v8bf*>(gA + k0 + 8);
        pb0 = *reinterpret_cast<const v8bf*>(gB + k0);
        pb1 = *reinterpret_cast<const v8bf*>(gB + k0 + 8);
    };
    auto stageTiles = [&]() {
        *reinterpret_cast<v8bf*>(sA)     = pa0;
        *reinterpret_cast<v8bf*>(sA + 8) = pa1;
        *reinterpret_cast<v8bf*>(sB)     = pb0;
        *reinterpret_cast<v8bf*>(sB + 8) = pb1;
    };

    v8f acc[2][4] = {};
    loadTiles(0);

    for (int k0 = 0; k0 < Kp; k0 += TILE_K) {
        stageTiles();
        __syncthreads();

        if (k0 + TILE_K < Kp) loadTiles(k0 + TILE_K);  // prefetch overlaps WMMAs

        v16bf af[2], bfr[4];
        #pragma unroll
        for (int i = 0; i < 2; ++i) {
            const __bf16* p = &As[(wm * 32 + i * 16 + l16) * LDSS + 8 * half];
            v8bf lo = *reinterpret_cast<const v8bf*>(p);
            v8bf hi = *reinterpret_cast<const v8bf*>(p + 16);
            af[i] = __builtin_shufflevector(lo, hi, 0,1,2,3,4,5,6,7,8,9,10,11,12,13,14,15);
        }
        #pragma unroll
        for (int j = 0; j < 4; ++j) {
            const __bf16* p = &Bs[(wn * 64 + j * 16 + l16) * LDSS + 8 * half];
            v8bf lo = *reinterpret_cast<const v8bf*>(p);
            v8bf hi = *reinterpret_cast<const v8bf*>(p + 16);
            bfr[j] = __builtin_shufflevector(lo, hi, 0,1,2,3,4,5,6,7,8,9,10,11,12,13,14,15);
        }
        #pragma unroll
        for (int i = 0; i < 2; ++i)
            #pragma unroll
            for (int j = 0; j < 4; ++j)
                acc[i][j] = __builtin_amdgcn_wmma_f32_16x16x32_bf16(
                    false, af[i], false, bfr[j], (short)0, acc[i][j], false, false);
        __syncthreads();
    }

    // ---- epilogue (M multiple of 128 -> no row guard) ----
    #pragma unroll
    for (int i = 0; i < 2; ++i) {
        #pragma unroll
        for (int j = 0; j < 4; ++j) {
            int col = col0 + wn * 64 + j * 16 + l16;
            if (col >= N) continue;
            float bv = bias[col];
            #pragma unroll
            for (int r = 0; r < 8; ++r) {
                int row = row0 + wm * 32 + i * 16 + r + 8 * half;
                float v = acc[i][j][r] + bv;
                size_t idx = (size_t)row * N + col;
                if (EPI == EPI_NONE) {
                    outf[idx] = v;
                    if (outb) outb[idx] = (__bf16)v;
                } else if (EPI == EPI_IND) {
                    outb[idx] = (__bf16)((v >= 1.0f) ? 1.0f : 0.0f);
                } else if (EPI == EPI_GELU_IND) {
                    outb[idx] = (__bf16)((v >= 1.0f) ? GELU1 : 0.0f);
                } else if (EPI == EPI_SIG_IND) {
                    outb[idx] = (__bf16)((v >= 1.0f) ? SIG1 : SIG0);
                } else if (EPI == EPI_SIGMOID) {
                    outf[idx] = 1.0f / (1.0f + __expf(-v));
                } else { // EPI_MASK_RESID: h = m * ind(v) + h ; also refresh bf16 mirror
                    float ind = (v >= 1.0f) ? 1.0f : 0.0f;
                    float hn  = (float)mbuf[idx] * ind + hbuf[idx];
                    outf[idx] = hn;
                    outb[idx] = (__bf16)hn;
                }
            }
        }
    }
}

// ---------------------------------------------------------------------------
// Weight prep: W fp32 [K][N] -> Wt bf16 [Np][Kp], transposed + zero padded.
// Classic 32x32 LDS-tiled transpose; grid covers the padded extent.
// ---------------------------------------------------------------------------
__global__ __launch_bounds__(256)
void wt_kernel(const float* __restrict__ W, __bf16* __restrict__ Wt,
               int K, int N, int Kp, int Np)
{
    __shared__ float tile[32][33];
    int kb = blockIdx.x * 32, nb = blockIdx.y * 32;
    int tx = threadIdx.x & 31, ty = threadIdx.x >> 5;   // 32 x 8
    #pragma unroll
    for (int i = ty; i < 32; i += 8) {
        int k = kb + i, n = nb + tx;
        tile[i][tx] = (k < K && n < N) ? W[(size_t)k * N + n] : 0.0f;
    }
    __syncthreads();
    #pragma unroll
    for (int i = ty; i < 32; i += 8) {
        int n = nb + i, k = kb + tx;
        if (n < Np && k < Kp) Wt[(size_t)n * Kp + k] = (__bf16)tile[tx][i];
    }
}

// X fp32 [256][36] -> Xb bf16 [256][64] zero-padded
__global__ void xconv_kernel(const float* __restrict__ X, __bf16* __restrict__ Xb)
{
    int idx = blockIdx.x * blockDim.x + threadIdx.x;    // 256*64
    if (idx >= 256 * 64) return;
    int r = idx >> 6, k = idx & 63;
    Xb[idx] = (__bf16)((k < 36) ? X[r * 36 + k] : 0.0f);
}

__global__ void zero_bf16_kernel(__bf16* __restrict__ p, int n)
{
    int idx = blockIdx.x * blockDim.x + threadIdx.x;
    if (idx < n) p[idx] = (__bf16)0.0f;
}

// ---------------------------------------------------------------------------
// LayerNorm: one wave per row (wave32 shuffle reduction); fp32 in, bf16 out.
// ---------------------------------------------------------------------------
__global__ __launch_bounds__(256)
void ln_kernel(const float* __restrict__ x, const float* __restrict__ g,
               const float* __restrict__ b, __bf16* __restrict__ y,
               int rows, int D)
{
    int wave = threadIdx.x >> 5;
    int lane = threadIdx.x & 31;
    int row  = blockIdx.x * 8 + wave;
    if (row >= rows) return;
    const float* xr = x + (size_t)row * D;
    float s = 0.0f, s2 = 0.0f;
    for (int d = lane; d < D; d += 32) { float v = xr[d]; s += v; s2 += v * v; }
    #pragma unroll
    for (int off = 16; off > 0; off >>= 1) {
        s  += __shfl_xor(s,  off, 32);
        s2 += __shfl_xor(s2, off, 32);
    }
    float m   = s / (float)D;
    float var = s2 / (float)D - m * m;
    float inv = rsqrtf(var + 1e-5f);
    __bf16* yr = y + (size_t)row * D;
    for (int d = lane; d < D; d += 32)
        yr[d] = (__bf16)((xr[d] - m) * inv * g[d] + b[d]);
}

// pooled[b,d] = mean_i x[b,i,d]   (bf16 in, bf16 out, fp32 accumulate)
__global__ void pool_kernel(const __bf16* __restrict__ x, __bf16* __restrict__ out,
                            int B, int S, int D)
{
    int idx = blockIdx.x * blockDim.x + threadIdx.x;
    if (idx >= B * D) return;
    int b = idx / D, d = idx % D;
    float s = 0.0f;
    for (int i = 0; i < S; ++i) s += (float)x[((size_t)b * S + i) * D + d];
    out[idx] = (__bf16)(s / (float)S);
}

// ---------------------------------------------------------------------------
// JAX threefry2x32 (20 rounds) + uniform->normal (matches jax.random.normal)
// ---------------------------------------------------------------------------
__device__ __forceinline__ void threefry2x32(uint32_t k0, uint32_t k1,
                                             uint32_t c0, uint32_t c1,
                                             uint32_t& o0, uint32_t& o1)
{
    uint32_t ks2 = k0 ^ k1 ^ 0x1BD11BDAu;
    uint32_t x0 = c0 + k0, x1 = c1 + k1;
#define TF_R(r) { x0 += x1; x1 = (x1 << (r)) | (x1 >> (32 - (r))); x1 ^= x0; }
    TF_R(13) TF_R(15) TF_R(26) TF_R(6)   x0 += k1;  x1 += ks2 + 1u;
    TF_R(17) TF_R(29) TF_R(16) TF_R(24)  x0 += ks2; x1 += k0  + 2u;
    TF_R(13) TF_R(15) TF_R(26) TF_R(6)   x0 += k0;  x1 += k1  + 3u;
    TF_R(17) TF_R(29) TF_R(16) TF_R(24)  x0 += k1;  x1 += ks2 + 4u;
    TF_R(13) TF_R(15) TF_R(26) TF_R(6)   x0 += ks2; x1 += k0  + 5u;
#undef TF_R
    o0 = x0; o1 = x1;
}

__device__ __forceinline__ void jax_split42(uint32_t& k1a, uint32_t& k1b,
                                            uint32_t& k2a, uint32_t& k2b)
{
    // key(42) = (0,42); split -> threefry(key, iota(4)) paired (0,2),(1,3)
    uint32_t a0, a1, b0, b1;
    threefry2x32(0u, 42u, 0u, 2u, a0, a1);
    threefry2x32(0u, 42u, 1u, 3u, b0, b1);
    k1a = a0; k1b = b0;   // nk1
    k2a = a1; k2b = b1;   // nk2
}

__device__ __forceinline__ float bits_to_normal(uint32_t bits)
{
    float u01 = __uint_as_float((bits >> 9) | 0x3f800000u) - 1.0f;  // [0,1)
    const float lo = -0.9999999403953552f;                          // nextafter(-1,0)
    float u = fmaxf(lo, u01 * (1.0f - lo) + lo);
    return 1.4142135623730951f * erfinvf(u);
}

// att[j] = sigmoid(mean_b aout[b][j]);   aout is 256x2 fp32
__global__ void att_kernel(const float* __restrict__ aout, float* __restrict__ att)
{
    __shared__ float s0[256], s1[256];
    int t = threadIdx.x;
    s0[t] = aout[t * 2 + 0];
    s1[t] = aout[t * 2 + 1];
    __syncthreads();
    for (int off = 128; off > 0; off >>= 1) {
        if (t < off) { s0[t] += s0[t + off]; s1[t] += s1[t + off]; }
        __syncthreads();
    }
    if (t == 0) {
        att[0] = 1.0f / (1.0f + __expf(-(s0[0] / 256.0f)));
        att[1] = 1.0f / (1.0f + __expf(-(s1[0] / 256.0f)));
    }
}

// coeff[b] = att0 + att1 * normal(nk1)[b];  B=256 -> 128 threefry pairs
__global__ void coeff_kernel(const float* __restrict__ att, float* __restrict__ coeff)
{
    uint32_t k1a, k1b, k2a, k2b;
    jax_split42(k1a, k1b, k2a, k2b);
    int i = threadIdx.x; // 0..127
    uint32_t o0, o1;
    threefry2x32(k1a, k1b, (uint32_t)i, (uint32_t)(i + 128), o0, o1);
    float a0 = att[0], a1 = att[1];
    coeff[i]       = a0 + a1 * bits_to_normal(o0);
    coeff[i + 128] = a0 + a1 * bits_to_normal(o1);
}

// grd[token=b*36+i][j] = (means[b,i] + stds[b,i]*normal(nk2)[b,i,j]) * coeff[b]
// written as bf16 with row stride 64 (cols 36..63 pre-zeroed)
__global__ void grid_kernel(const float* __restrict__ distr,
                            const float* __restrict__ coeff,
                            __bf16* __restrict__ grd)
{
    const int total = 256 * 36 * 36;
    const int halfn = total / 2;
    int p = blockIdx.x * blockDim.x + threadIdx.x;
    if (p >= halfn) return;
    uint32_t k1a, k1b, k2a, k2b;
    jax_split42(k1a, k1b, k2a, k2b);
    uint32_t o0, o1;
    threefry2x32(k2a, k2b, (uint32_t)p, (uint32_t)(p + halfn), o0, o1);
    #pragma unroll
    for (int h = 0; h < 2; ++h) {
        int f = p + h * halfn;
        float z = bits_to_normal(h ? o1 : o0);
        int b   = f / 1296;
        int rem = f - b * 1296;
        int i   = rem / 36;
        int j   = rem - i * 36;
        float mean = distr[b * 72 + 2 * i];
        float sd   = distr[b * 72 + 2 * i + 1];
        grd[(size_t)(b * 36 + i) * 64 + j] = (__bf16)((mean + sd * z) * coeff[b]);
    }
}

// ---------------------------------------------------------------------------
// Host-side orchestration
// ---------------------------------------------------------------------------
static void launch_gemm(int epi, const __bf16* A, const __bf16* Wt, const float* b,
                        float* outf, __bf16* outb, int M, int Kp, int N,
                        const __bf16* mbuf, const float* hbuf, hipStream_t s)
{
    dim3 grid((M + TILE_M - 1) / TILE_M, (N + TILE_N - 1) / TILE_N);
    dim3 blk(256);
    switch (epi) {
    case 0: wmma_gemm_kernel<0><<<grid, blk, 0, s>>>(A, Wt, b, outf, outb, M, Kp, N, mbuf, hbuf); break;
    case 1: wmma_gemm_kernel<1><<<grid, blk, 0, s>>>(A, Wt, b, outf, outb, M, Kp, N, mbuf, hbuf); break;
    case 2: wmma_gemm_kernel<2><<<grid, blk, 0, s>>>(A, Wt, b, outf, outb, M, Kp, N, mbuf, hbuf); break;
    case 3: wmma_gemm_kernel<3><<<grid, blk, 0, s>>>(A, Wt, b, outf, outb, M, Kp, N, mbuf, hbuf); break;
    case 4: wmma_gemm_kernel<4><<<grid, blk, 0, s>>>(A, Wt, b, outf, outb, M, Kp, N, mbuf, hbuf); break;
    default: wmma_gemm_kernel<5><<<grid, blk, 0, s>>>(A, Wt, b, outf, outb, M, Kp, N, mbuf, hbuf); break;
    }
}

extern "C" void kernel_launch(void* const* d_in, const int* in_sizes, int n_in,
                              void* d_out, int out_size, void* d_ws, size_t ws_size,
                              hipStream_t stream)
{
    (void)in_sizes; (void)out_size; (void)ws_size;
    if (n_in < 259) return;

    const int B = 256, S = 36, D = 512, DEPTH = 12;
    const int M = B * S;                       // 9216 tokens

    // ---- input pointer map (setup_inputs() dict insertion order) ----
    #define FP(i) ((const float*)d_in[(i)])
    const float* X = FP(0);
    // lD: 1..6, lA: 7..12, embed: 13..14, blocks: 15 + blk*20, final ln: 255,256, out: 257,258

    // ---- workspace carve-up ----
    char* ws = (char*)d_ws;
    size_t off = 0;
    auto carveB = [&](size_t bytes) { void* p = ws + off; off += (bytes + 255) & ~(size_t)255; return p; };
    auto carveF = [&](size_t elems) { return (float*) carveB(elems * 4); };
    auto carveH = [&](size_t elems) { return (__bf16*)carveB(elems * 2); };

    float*  h     = carveF((size_t)M * D);        // residual stream (fp32)
    __bf16* hbf   = carveH((size_t)M * D);        // bf16 mirror of h
    __bf16* lnb   = carveH((size_t)M * D);        // LN output
    __bf16* m1    = carveH((size_t)M * D);        // mask hidden
    __bf16* mbuf  = carveH((size_t)M * D);        // mask gate m
    __bf16* big   = carveH((size_t)M * 4 * D);    // ff hidden (up to 4D)
    __bf16* grd   = carveH((size_t)M * 64);       // grid, row stride 64 (Kp of 36)
    __bf16* Xb    = carveH((size_t)B * 64);       // bf16 X, row stride 64
    __bf16* g1    = carveH((size_t)B * D);
    __bf16* g2    = carveH((size_t)B * D);
    float*  distr = carveF((size_t)B * 72);
    float*  aout  = carveF((size_t)B * 2);
    float*  att   = carveF(2);
    float*  coeff = carveF((size_t)B);
    __bf16* pool  = carveH((size_t)B * D);

    // weight prep (lazy, on-stream): W fp32 [K][N] -> Wt bf16 [Np][Kp]
    auto prepWt = [&](int wi, int K, int N) -> const __bf16* {
        int Kp = ceil32(K), Np = ceil128(N);
        __bf16* wt = carveH((size_t)Np * Kp);
        dim3 g(Kp / 32, Np / 32), blk(256);
        wt_kernel<<<g, blk, 0, stream>>>(FP(wi), wt, K, N, Kp, Np);
        return wt;
    };

    // ---- X -> bf16 (padded) ----
    xconv_kernel<<<(256 * 64 + 255) / 256, 256, 0, stream>>>(X, Xb);

    // ---- head3 lD -> distr = sigmoid(...) ----
    launch_gemm(EPI_IND,     Xb, prepWt(1, S, D), FP(2), nullptr, g1,    B, 64,  D,  nullptr, nullptr, stream);
    launch_gemm(EPI_IND,     g1, prepWt(3, D, D), FP(4), nullptr, g2,    B, D,   D,  nullptr, nullptr, stream);
    launch_gemm(EPI_SIGMOID, g2, prepWt(5, D, 72), FP(6), distr, nullptr, B, D,  72, nullptr, nullptr, stream);

    // ---- head3 lA -> aout (256x2) -> att (2) -> coeff (256) ----
    launch_gemm(EPI_IND,  Xb, prepWt(7, S, D), FP(8),  nullptr, g1,   B, 64, D, nullptr, nullptr, stream);
    launch_gemm(EPI_IND,  g1, prepWt(9, D, D), FP(10), nullptr, g2,   B, D,  D, nullptr, nullptr, stream);
    launch_gemm(EPI_NONE, g2, prepWt(11, D, 2), FP(12), aout, nullptr, B, D,  2, nullptr, nullptr, stream);
    att_kernel<<<1, 256, 0, stream>>>(aout, att);
    coeff_kernel<<<1, 128, 0, stream>>>(att, coeff);

    // ---- grid noise (bf16, padded stride 64) + embed: h = grid @ We + be ----
    zero_bf16_kernel<<<((int)(M * 64) + 255) / 256, 256, 0, stream>>>(grd, M * 64);
    grid_kernel<<<(B * S * S / 2 + 255) / 256, 256, 0, stream>>>(distr, coeff, grd);
    launch_gemm(EPI_NONE, grd, prepWt(13, S, D), FP(14), h, hbf, M, 64, D, nullptr, nullptr, stream);

    // ---- 12 blocks x 2 pre-norm-residual units ----
    for (int blk = 0; blk < DEPTH; ++blk) {
        for (int pnr = 0; pnr < 2; ++pnr) {
            int base = 15 + blk * 20 + pnr * 10;
            int dff  = (pnr == 0) ? 4 * D : D / 2;
            // m = sigmoid(ind(gelu(ind(h@Wm0+b))@Wm1+b))  -> bf16 gate
            launch_gemm(EPI_GELU_IND, hbf, prepWt(base + 2, D, D), FP(base + 3), nullptr, m1,   M, D, D, nullptr, nullptr, stream);
            launch_gemm(EPI_SIG_IND,  m1,  prepWt(base + 4, D, D), FP(base + 5), nullptr, mbuf, M, D, D, nullptr, nullptr, stream);
            // y path: LN -> gelu(ind) -> ind, fused residual: h = m*ind(...) + h
            ln_kernel<<<(M + 7) / 8, 256, 0, stream>>>(h, FP(base + 0), FP(base + 1), lnb, M, D);
            launch_gemm(EPI_GELU_IND,   lnb, prepWt(base + 6, D, dff), FP(base + 7), nullptr, big, M, D,   dff, nullptr, nullptr, stream);
            launch_gemm(EPI_MASK_RESID, big, prepWt(base + 8, dff, D), FP(base + 9), h,       hbf, M, dff, D,   mbuf, h, stream);
        }
    }

    // ---- final LN, seq-mean, output head ----
    ln_kernel<<<(M + 7) / 8, 256, 0, stream>>>(h, FP(255), FP(256), lnb, M, D);
    pool_kernel<<<(B * D + 255) / 256, 256, 0, stream>>>(lnb, pool, B, S, D);
    launch_gemm(EPI_NONE, pool, prepWt(257, D, 68), FP(258), (float*)d_out, nullptr, B, D, 68, nullptr, nullptr, stream);
    #undef FP
}
